// DeformConv2d_for_Style_75591424409914
// MI455X (gfx1250) — compile-verified, matrix operand order checked by reading, not run on hardware
//
#include <hip/hip_runtime.h>
#include <stdint.h>

// ---------------------------------------------------------------------------
// DeformConv2d (offset conv + bilinear im2col + bf16 WMMA GEMM) for gfx1250
// B=8, CIN=128, COUT=256, K=3, S=1, P=1, H=W=64  ->  M=32768, N=256, Kdim=1152
// ---------------------------------------------------------------------------

#define BATCH 8
#define CINC  128
#define COUTC 256
#define HWC   4096      // 64*64
#define KDIM  1152      // 9*128, reduction index k = kk*128 + c
#define MTOT  32768     // BATCH*HWC

typedef __bf16 bf16_t;
typedef __attribute__((ext_vector_type(16))) bf16_t v16bf;
typedef __attribute__((ext_vector_type(8)))  float  v8f;

union Frag { uint4 u[2]; v16bf v; };

static __device__ __forceinline__ unsigned short f32_to_bf16(float f) {
    union { float f; unsigned u; } v; v.f = f;
    unsigned r = v.u + 0x7FFFu + ((v.u >> 16) & 1u);   // round-to-nearest-even
    return (unsigned short)(r >> 16);
}

static __device__ __forceinline__ float tap(const float* __restrict__ xb, int yy, int xx) {
    // reference: clamp index, multiply by validity -> equivalent to 0 outside
    if ((unsigned)yy < 64u && (unsigned)xx < 64u) return xb[(yy << 6) + xx];
    return 0.0f;
}

// ---------------------------------------------------------------------------
// Kernel 1: offset-predicting conv, 128->18 ch, 3x3, reflect pad (reflect_101)
// one thread per (b, pixel); weights are thread-uniform -> scalar loads
// ---------------------------------------------------------------------------
__global__ __launch_bounds__(256) void k_offconv(
    const float* __restrict__ x, const float* __restrict__ w_off,
    const float* __restrict__ b_off, float* __restrict__ off)
{
    int m = blockIdx.x * 256 + threadIdx.x;
    if (m >= MTOT) return;
    int b = m >> 12, l = m & 4095, y = l >> 6, xx = l & 63;

    float acc[18];
#pragma unroll
    for (int oc = 0; oc < 18; ++oc) acc[oc] = b_off[oc];

    const float* xb = x + (size_t)b * CINC * HWC;
    for (int c = 0; c < CINC; ++c) {
        float xv[9];
#pragma unroll
        for (int i = 0; i < 3; ++i) {
            int ry = y - 1 + i; ry = ry < 0 ? -ry : (ry > 63 ? 126 - ry : ry);
#pragma unroll
            for (int j = 0; j < 3; ++j) {
                int rx = xx - 1 + j; rx = rx < 0 ? -rx : (rx > 63 ? 126 - rx : rx);
                xv[i * 3 + j] = xb[c * HWC + (ry << 6) + rx];
            }
        }
        const float* wr = w_off + c * 9;          // w_off[oc][c][t] = oc*1152 + c*9 + t
#pragma unroll
        for (int oc = 0; oc < 18; ++oc) {
#pragma unroll
            for (int t = 0; t < 9; ++t) acc[oc] += wr[oc * KDIM + t] * xv[t];
        }
    }
    float* ob = off + (size_t)b * 18 * HWC + l;
#pragma unroll
    for (int oc = 0; oc < 18; ++oc) ob[oc * HWC] = acc[oc];
}

// ---------------------------------------------------------------------------
// Kernel 2: permute + convert weights to bf16:  WB[n][kk*128+c] = w_dc[n][c][kk]
// ---------------------------------------------------------------------------
__global__ __launch_bounds__(256) void k_wcvt(
    const float* __restrict__ w_dc, unsigned short* __restrict__ WB)
{
    int idx = blockIdx.x * 256 + threadIdx.x;
    if (idx >= COUTC * KDIM) return;
    int n = idx / KDIM, k = idx - n * KDIM;
    int kk = k >> 7, c = k & 127;
    WB[idx] = f32_to_bf16(w_dc[n * KDIM + c * 9 + kk]);
}

// ---------------------------------------------------------------------------
// Kernel 3: bilinear gather -> im2col (bf16).  One block per pixel m,
// one thread per input channel c; offsets are block-uniform (scalar loads);
// writes of 128 contiguous bf16 per kk -> fully coalesced.  x (16.8 MB) is
// L2-resident (192 MB L2), so the strided 4-corner taps are L2 hits.
// ---------------------------------------------------------------------------
__global__ __launch_bounds__(128) void k_cols(
    const float* __restrict__ x, const float* __restrict__ off,
    unsigned short* __restrict__ cols)
{
    int m = blockIdx.x;
    int c = threadIdx.x;
    int b = m >> 12, l = m & 4095, y = l >> 6, xx = l & 63;

    const float* xb = x + (size_t)(b * CINC + c) * HWC;
    const float* ob = off + (size_t)b * 18 * HWC + l;
    unsigned short* crow = cols + (size_t)m * KDIM;

#pragma unroll
    for (int kk = 0; kk < 9; ++kk) {
        int ki = kk / 3, kj = kk % 3;
        float py = ob[(kk * 2 + 0) * HWC] + (float)(y - 1 + ki);
        float px = ob[(kk * 2 + 1) * HWC] + (float)(xx - 1 + kj);
        float fy0 = floorf(py), fx0 = floorf(px);
        int y0 = (int)fy0, x0 = (int)fx0;
        float dy = py - fy0, dx = px - fx0;
        float v00 = tap(xb, y0,     x0    );
        float v01 = tap(xb, y0,     x0 + 1);
        float v10 = tap(xb, y0 + 1, x0    );
        float v11 = tap(xb, y0 + 1, x0 + 1);
        float val = v00 * (1.f - dy) * (1.f - dx) + v01 * (1.f - dy) * dx
                  + v10 * dy * (1.f - dx) + v11 * dy * dx;
        crow[kk * 128 + c] = f32_to_bf16(val);
    }
}

// ---------------------------------------------------------------------------
// Kernel 4: GEMM  out[m][n] = A[m][:] . WB[n][:]  via v_wmma_f32_16x16x32_bf16
// Wave tile = 64(M) x 64(N): 4x4 = 16 f32 accumulators (128 VGPRs).
// Per k-step: 4 A-frags + 4 B-frags (16 independent global_load_b128 -> one
// clause, one wait), then a 16-deep back-to-back v_wmma burst reusing them
// (wmma:load ratio 16:16 vs 16:34 of the 16x256 tiling).  Block = 8 waves
// arranged 2(M) x 4(N) over a 128x256 tile, so waves sharing an M-strip hit
// the same A lines in WGP$/L2 -> A still streams from HBM only ~once.
// ---------------------------------------------------------------------------
__global__ __launch_bounds__(256) void k_gemm(
    const unsigned short* __restrict__ A, const unsigned short* __restrict__ WB,
    const float* __restrict__ bias, float* __restrict__ out)
{
    int lane  = threadIdx.x & 31;
    int wave  = threadIdx.x >> 5;
    int wm    = wave >> 2;                          // 0..1  (M strip in block)
    int wn    = wave & 3;                           // 0..3  (N strip in block)
    int m0    = blockIdx.x * 128 + wm * 64;
    int n0    = wn * 64;
    int ln15  = lane & 15;
    int half8 = (lane >> 4) << 3;                   // 0 or 8 (k offset)

    const unsigned short* arow = A  + (size_t)(m0 + ln15) * KDIM + half8;
    const unsigned short* brow = WB + (size_t)(n0 + ln15) * KDIM + half8;

    v8f acc[4][4];
#pragma unroll
    for (int mt = 0; mt < 4; ++mt)
#pragma unroll
        for (int nt = 0; nt < 4; ++nt)
            acc[mt][nt] = (v8f){0.f,0.f,0.f,0.f,0.f,0.f,0.f,0.f};

    for (int kb = 0; kb < KDIM; kb += 32) {
        Frag af[4], bf[4];
#pragma unroll
        for (int t = 0; t < 4; ++t) {
            const unsigned short* ap = arow + (size_t)t * 16 * KDIM + kb;
            af[t].u[0] = *reinterpret_cast<const uint4*>(ap);
            af[t].u[1] = *reinterpret_cast<const uint4*>(ap + 16);
        }
#pragma unroll
        for (int t = 0; t < 4; ++t) {
            const unsigned short* bp = brow + (size_t)t * 16 * KDIM + kb;
            bf[t].u[0] = *reinterpret_cast<const uint4*>(bp);
            bf[t].u[1] = *reinterpret_cast<const uint4*>(bp + 16);
        }
        if (kb + 64 < KDIM) {
#pragma unroll
            for (int t = 0; t < 4; ++t)                        // global_prefetch_b8
                __builtin_prefetch(arow + (size_t)t * 16 * KDIM + kb + 64, 0, 1);
        }
#pragma unroll
        for (int mt = 0; mt < 4; ++mt)
#pragma unroll
            for (int nt = 0; nt < 4; ++nt)
                acc[mt][nt] = __builtin_amdgcn_wmma_f32_16x16x32_bf16(
                    false, af[mt].v, false, bf[nt].v, (short)0, acc[mt][nt],
                    false, false);
    }

    // C/D layout: VGPR r -> row rm + r + (lane>=16 ? 8 : 0), col = nt*16 + (lane&15)
#pragma unroll
    for (int mt = 0; mt < 4; ++mt) {
        int rm = m0 + mt * 16;
        int b  = rm >> 12;
        int l0 = (rm & 4095) + half8;               // 8 consecutive pixels per lane
#pragma unroll
        for (int nt = 0; nt < 4; ++nt) {
            int oc = n0 + nt * 16 + ln15;
            float bi = bias[oc];
            float* o = out + ((size_t)(b * COUTC + oc) << 12) + l0;
            float4 lo, hi;
            lo.x = acc[mt][nt][0] + bi; lo.y = acc[mt][nt][1] + bi;
            lo.z = acc[mt][nt][2] + bi; lo.w = acc[mt][nt][3] + bi;
            hi.x = acc[mt][nt][4] + bi; hi.y = acc[mt][nt][5] + bi;
            hi.z = acc[mt][nt][6] + bi; hi.w = acc[mt][nt][7] + bi;
            *reinterpret_cast<float4*>(o)     = lo;
            *reinterpret_cast<float4*>(o + 4) = hi;
        }
    }
}

// ---------------------------------------------------------------------------
extern "C" void kernel_launch(void* const* d_in, const int* in_sizes, int n_in,
                              void* d_out, int out_size, void* d_ws, size_t ws_size,
                              hipStream_t stream)
{
    const float* x     = (const float*)d_in[0];   // [8,128,64,64]
    const float* w_off = (const float*)d_in[1];   // [18,128,3,3]
    const float* b_off = (const float*)d_in[2];   // [18]
    const float* w_dc  = (const float*)d_in[3];   // [256,128,3,3]
    const float* b_dc  = (const float*)d_in[4];   // [256]
    float* out = (float*)d_out;                   // [8,256,64,64]

    char* ws = (char*)d_ws;
    unsigned short* WB   = (unsigned short*)ws;                       //   589,824 B
    float*          OFF  = (float*)(ws + 589824);                     // 2,359,296 B
    unsigned short* COLS = (unsigned short*)(ws + 589824 + 2359296);  // 75,497,472 B
    (void)ws_size; (void)in_sizes; (void)n_in; (void)out_size;

    k_offconv<<<MTOT / 256, 256, 0, stream>>>(x, w_off, b_off, OFF);
    k_wcvt<<<(COUTC * KDIM + 255) / 256, 256, 0, stream>>>(w_dc, WB);
    k_cols<<<MTOT, 128, 0, stream>>>(x, OFF, COLS);
    k_gemm<<<MTOT / 128, 256, 0, stream>>>(COLS, WB, b_dc, out);
}
